// VQVariationalAutoEncoder_24343874634364
// MI455X (gfx1250) — compile-verified, matrix operand order checked by reading, not run on hardware
//
#include <hip/hip_runtime.h>
#include <math.h>

// ---------------------------------------------------------------------------
// Types for CDNA5 WMMA (wave32, V_WMMA_F32_16X16X32_BF16)
// ---------------------------------------------------------------------------
typedef __bf16 bf16_t;
typedef __attribute__((ext_vector_type(16))) __bf16 v16bf;
typedef __attribute__((ext_vector_type(8)))  __bf16 v8bf;
typedef __attribute__((ext_vector_type(8)))  float  v8f;

union Frag16 { v16bf v; v8bf h[2]; };

// ---------------------------------------------------------------------------
// im2col for normal conv.  dst: bf16 [M, Kpad], M = B*Ho*Wo, k=(ci*KH+r)*KW+s
// srcNCHW=1 -> src is NCHW f32, else NHWC f32.  relu applied on read.
// ---------------------------------------------------------------------------
__global__ void k_im2col_conv(const float* __restrict__ src, bf16_t* __restrict__ dst,
                              int Bn, int Cin, int Hin, int Win,
                              int KH, int KW, int stride, int pad,
                              int Ho, int Wo, int K, int Kpad,
                              int srcNCHW, int relu)
{
    size_t idx = (size_t)blockIdx.x * blockDim.x + threadIdx.x;
    size_t total = (size_t)Bn * Ho * Wo * Kpad;
    if (idx >= total) return;
    int k = (int)(idx % Kpad);
    size_t m = idx / Kpad;
    float v = 0.0f;
    if (k < K) {
        int s  = k % KW;
        int r  = (k / KW) % KH;
        int ci = k / (KW * KH);
        int wo = (int)(m % Wo);
        int ho = (int)((m / Wo) % Ho);
        int b  = (int)(m / ((size_t)Wo * Ho));
        int hi = ho * stride - pad + r;
        int wi = wo * stride - pad + s;
        if (hi >= 0 && hi < Hin && wi >= 0 && wi < Win) {
            size_t off = srcNCHW
                ? ((((size_t)b * Cin + ci) * Hin + hi) * Win + wi)
                : ((((size_t)b * Hin + hi) * Win + wi) * Cin + ci);
            v = src[off];
            if (relu && v < 0.0f) v = 0.0f;
        }
    }
    dst[idx] = (bf16_t)v;
}

// ---------------------------------------------------------------------------
// im2col for ConvTranspose2d(k=4,s=2,p=1), one output-parity class (ph,pw).
//   hi = hq + ph - i, wi = wq + pw - j   (i,j in {0,1}), k = ci*4 + i*2 + j.
// ---------------------------------------------------------------------------
__global__ void k_im2col_deconv(const float* __restrict__ src, bf16_t* __restrict__ dst,
                                int Bn, int Cin, int Hq, int Wq, int ph, int pw)
{
    int Kpad = Cin * 4;
    size_t idx = (size_t)blockIdx.x * blockDim.x + threadIdx.x;
    size_t total = (size_t)Bn * Hq * Wq * Kpad;
    if (idx >= total) return;
    int k = (int)(idx % Kpad);
    size_t m = idx / Kpad;
    int ci = k >> 2;
    int i  = (k >> 1) & 1;
    int j  = k & 1;
    int wq = (int)(m % Wq);
    int hq = (int)((m / Wq) % Hq);
    int b  = (int)(m / ((size_t)Wq * Hq));
    int hi = hq + ph - i;
    int wi = wq + pw - j;
    float v = 0.0f;
    if (hi >= 0 && hi < Hq && wi >= 0 && wi < Wq)
        v = src[(((size_t)b * Hq + hi) * Wq + wi) * Cin + ci];
    dst[idx] = (bf16_t)v;
}

// ---------------------------------------------------------------------------
// Weight packs (f32 -> bf16 Bt[Npad, Kpad], row-major over K)
// ---------------------------------------------------------------------------
__global__ void k_pack_w_conv(const float* __restrict__ w, bf16_t* __restrict__ Bt,
                              int Cout, int K, int Kpad, int Npad)
{
    size_t idx = (size_t)blockIdx.x * blockDim.x + threadIdx.x;
    size_t total = (size_t)Npad * Kpad;
    if (idx >= total) return;
    int k = (int)(idx % Kpad);
    int n = (int)(idx / Kpad);
    float v = (n < Cout && k < K) ? w[(size_t)n * K + k] : 0.0f;
    Bt[idx] = (bf16_t)v;
}

__global__ void k_pack_w_deconv(const float* __restrict__ w, bf16_t* __restrict__ Bt,
                                int Cin, int Cout, int Npad, int ph, int pw)
{
    int Kpad = Cin * 4;
    size_t idx = (size_t)blockIdx.x * blockDim.x + threadIdx.x;
    size_t total = (size_t)Npad * Kpad;
    if (idx >= total) return;
    int k = (int)(idx % Kpad);
    int n = (int)(idx / Kpad);
    float v = 0.0f;
    if (n < Cout) {
        int ci = k >> 2;
        int i  = (k >> 1) & 1;
        int j  = k & 1;
        int kh = (1 - ph) + 2 * i;
        int kw = (1 - pw) + 2 * j;
        v = w[(((size_t)ci * Cout + n) * 4 + kh) * 4 + kw];
    }
    Bt[idx] = (bf16_t)v;
}

// ---------------------------------------------------------------------------
// Shared epilogue helper (device inline): writes one 16x16 tile's lane data.
// ---------------------------------------------------------------------------
__device__ __forceinline__ void epilogue_store(
    const v8f& c, float* __restrict__ out,
    const float* __restrict__ bias, const float* __restrict__ res,
    int mBase, int nBase, int half, int lr,
    int N, int mode, int Hq, int Wq, int Ho, int Wo,
    int ph, int pw, int CoutReal)
{
    int n = nBase + lr;
    #pragma unroll
    for (int i = 0; i < 8; ++i) {
        int m = mBase + half * 8 + i;
        float acc = c[i];
        if (mode == 0) {
            size_t o = (size_t)m * N + n;
            if (bias) acc += bias[n];
            if (res)  acc += res[o];
            out[o] = acc;
        } else {
            int b  = m / (Hq * Wq);
            int rm = m - b * Hq * Wq;
            int hq = rm / Wq;
            int wq = rm - hq * Wq;
            int ho = 2 * hq + ph;
            int wo = 2 * wq + pw;
            if (mode == 1) {
                size_t o = ((((size_t)b * Ho + ho) * Wo + wo)) * N + n;
                out[o] = acc + bias[n];
            } else {
                if (n < CoutReal) {
                    size_t o = (((size_t)b * CoutReal + n) * Ho + ho) * Wo + wo;
                    out[o] = tanhf(acc + bias[n]);
                }
            }
        }
    }
}

// ---------------------------------------------------------------------------
// Hierarchically blocked GEMM: C[M,N] = A[M,K] x Bt[N,K]^T  (bf16 in, f32 acc)
//   - block macro-tile 128x128, K-step 32, 256 threads = 8 waves
//   - A/B K-slices double-buffered in LDS (row pitch 40 bf16 = 80B: the 16
//     fragment rows hit disjoint 4-bank groups -> conflict-free ds_load_b128)
//   - each wave computes 2(M) x 4(N) WMMA tiles reusing LDS fragments
//   - next K-slice is register-staged from global during compute
// Global traffic: 16KB per 1.05 MFLOP macro-step  ->  64 FLOP/byte.
// Requires M%128==0, N%128==0, K%32==0.
// ---------------------------------------------------------------------------
#define LDSP 40  // padded row pitch in bf16

__global__ __launch_bounds__(256) void k_gemm_tiled_wmma(
    const bf16_t* __restrict__ A, const bf16_t* __restrict__ Bt,
    float* __restrict__ out, const float* __restrict__ bias,
    const float* __restrict__ res,
    int M, int N, int Kpad, int mode,
    int Hq, int Wq, int Ho, int Wo, int ph, int pw, int CoutReal)
{
    __shared__ __align__(16) bf16_t Alds[2][128 * LDSP];
    __shared__ __align__(16) bf16_t Blds[2][128 * LDSP];

    const int tilesN = N >> 7;
    const int bm = blockIdx.x / tilesN;
    const int bn = blockIdx.x % tilesN;
    const int mBlock = bm << 7;
    const int nBlock = bn << 7;

    const int t    = (int)threadIdx.x;
    const int w    = t >> 5;
    const int lane = t & 31;
    const int half = lane >> 4;
    const int lr   = lane & 15;
    const int wm   = w & 3;   // 4 waves along M, 32 rows each
    const int wn   = w >> 2;  // 2 waves along N, 64 cols each

    // Cooperative staging: 512 16B chunks per matrix per K-step; 2 per thread.
    // chunk c: row = c>>2 (0..127), kc = (c&3)*8
    const int r0 = t >> 2,        kc0 = (t & 3) << 3;
    const int r1 = (t + 256) >> 2, kc1 = ((t + 256) & 3) << 3;

    const bf16_t* gA0 = A  + (size_t)(mBlock + r0) * Kpad + kc0;
    const bf16_t* gA1 = A  + (size_t)(mBlock + r1) * Kpad + kc1;
    const bf16_t* gB0 = Bt + (size_t)(nBlock + r0) * Kpad + kc0;
    const bf16_t* gB1 = Bt + (size_t)(nBlock + r1) * Kpad + kc1;

    v8bf ra0, ra1, rb0, rb1;

    // preload K-slice 0
    ra0 = *(const v8bf*)(gA0);
    ra1 = *(const v8bf*)(gA1);
    rb0 = *(const v8bf*)(gB0);
    rb1 = *(const v8bf*)(gB1);
    *(v8bf*)&Alds[0][r0 * LDSP + kc0] = ra0;
    *(v8bf*)&Alds[0][r1 * LDSP + kc1] = ra1;
    *(v8bf*)&Blds[0][r0 * LDSP + kc0] = rb0;
    *(v8bf*)&Blds[0][r1 * LDSP + kc1] = rb1;
    __syncthreads();

    v8f acc[2][4];
    #pragma unroll
    for (int mi = 0; mi < 2; ++mi)
        #pragma unroll
        for (int nj = 0; nj < 4; ++nj)
            acc[mi][nj] = (v8f){0.f,0.f,0.f,0.f,0.f,0.f,0.f,0.f};

    int p = 0;
    for (int k0 = 0; k0 < Kpad; k0 += 32) {
        const bool more = (k0 + 32) < Kpad;
        if (more) {  // register-stage next K-slice while computing
            ra0 = *(const v8bf*)(gA0 + k0 + 32);
            ra1 = *(const v8bf*)(gA1 + k0 + 32);
            rb0 = *(const v8bf*)(gB0 + k0 + 32);
            rb1 = *(const v8bf*)(gB1 + k0 + 32);
        }

        Frag16 fa[2], fb[4];
        #pragma unroll
        for (int mi = 0; mi < 2; ++mi) {
            int row = wm * 32 + mi * 16 + lr;
            fa[mi].h[0] = *(const v8bf*)&Alds[p][row * LDSP + half * 8];
            fa[mi].h[1] = *(const v8bf*)&Alds[p][row * LDSP + 16 + half * 8];
        }
        #pragma unroll
        for (int nj = 0; nj < 4; ++nj) {
            int row = wn * 64 + nj * 16 + lr;
            fb[nj].h[0] = *(const v8bf*)&Blds[p][row * LDSP + half * 8];
            fb[nj].h[1] = *(const v8bf*)&Blds[p][row * LDSP + 16 + half * 8];
        }
        #pragma unroll
        for (int mi = 0; mi < 2; ++mi)
            #pragma unroll
            for (int nj = 0; nj < 4; ++nj)
                acc[mi][nj] = __builtin_amdgcn_wmma_f32_16x16x32_bf16(
                    false, fa[mi].v, false, fb[nj].v, (short)0,
                    acc[mi][nj], false, false);

        if (more) {
            *(v8bf*)&Alds[p ^ 1][r0 * LDSP + kc0] = ra0;
            *(v8bf*)&Alds[p ^ 1][r1 * LDSP + kc1] = ra1;
            *(v8bf*)&Blds[p ^ 1][r0 * LDSP + kc0] = rb0;
            *(v8bf*)&Blds[p ^ 1][r1 * LDSP + kc1] = rb1;
        }
        __syncthreads();
        p ^= 1;
    }

    #pragma unroll
    for (int mi = 0; mi < 2; ++mi)
        #pragma unroll
        for (int nj = 0; nj < 4; ++nj)
            epilogue_store(acc[mi][nj], out, bias, res,
                           mBlock + wm * 32 + mi * 16,
                           nBlock + wn * 64 + nj * 16,
                           half, lr, N, mode, Hq, Wq, Ho, Wo, ph, pw, CoutReal);
}

// ---------------------------------------------------------------------------
// Simple one-tile-per-wave GEMM (used only for N=16 final deconv, ~1% FLOPs)
// ---------------------------------------------------------------------------
__global__ __launch_bounds__(256) void k_gemm_bf16_wmma(
    const bf16_t* __restrict__ A, const bf16_t* __restrict__ Bt,
    float* __restrict__ out, const float* __restrict__ bias,
    const float* __restrict__ res,
    int M, int N, int Kpad, int mode,
    int Hq, int Wq, int Ho, int Wo, int ph, int pw, int CoutReal)
{
    const int tilesN = N >> 4;
    const int tilesM = M >> 4;
    const int totalT = tilesM * tilesN;
    int wave = blockIdx.x * (blockDim.x >> 5) + ((int)threadIdx.x >> 5);
    if (wave >= totalT) return;
    int tm = wave / tilesN;
    int tn = wave % tilesN;
    int mBase = tm << 4;
    int nBase = tn << 4;
    int lane = (int)threadIdx.x & 31;
    int half = lane >> 4;
    int lr   = lane & 15;

    const bf16_t* aptr = A  + (size_t)(mBase + lr) * Kpad + half * 8;
    const bf16_t* bptr = Bt + (size_t)(nBase + lr) * Kpad + half * 8;

    v8f c = {0.f, 0.f, 0.f, 0.f, 0.f, 0.f, 0.f, 0.f};
    for (int k0 = 0; k0 < Kpad; k0 += 32) {
        Frag16 fa, fb;
        fa.h[0] = *(const v8bf*)(aptr + k0);
        fa.h[1] = *(const v8bf*)(aptr + k0 + 16);
        fb.h[0] = *(const v8bf*)(bptr + k0);
        fb.h[1] = *(const v8bf*)(bptr + k0 + 16);
        c = __builtin_amdgcn_wmma_f32_16x16x32_bf16(
            false, fa.v, false, fb.v, (short)0, c, false, false);
    }
    epilogue_store(c, out, bias, res, mBase, nBase, half, lr,
                   N, mode, Hq, Wq, Ho, Wo, ph, pw, CoutReal);
}

// ---------------------------------------------------------------------------
// VQ: z in-place, NHWC f32 viewed as [Nrows, 64] rows, codebook in 64KB LDS.
// ---------------------------------------------------------------------------
__global__ __launch_bounds__(256) void k_vq(float* __restrict__ z,
                                            const float* __restrict__ codebook,
                                            float* __restrict__ loss,
                                            int Nrows)
{
    __shared__ float cb[256 * 64];
    for (int i = threadIdx.x; i < 256 * 64; i += blockDim.x)
        cb[i] = codebook[i];
    __syncthreads();

    int row = blockIdx.x * blockDim.x + threadIdx.x;
    if (row >= Nrows) return;

    float zr[64];
    float* zp = z + (size_t)row * 64;
    #pragma unroll
    for (int j = 0; j < 64; ++j) zr[j] = zp[j];

    float best = 3.4e38f;
    int   bi   = 0;
    for (int e = 0; e < 256; ++e) {
        const float* er = &cb[e * 64];
        float d = 0.f;
        #pragma unroll
        for (int j = 0; j < 64; ++j) {
            float t2 = zr[j] - er[j];
            d = fmaf(t2, t2, d);
        }
        if (d < best) { best = d; bi = e; }
    }
    const float* er = &cb[bi * 64];
    #pragma unroll
    for (int j = 0; j < 64; ++j) zp[j] = er[j];

    atomicAdd(loss, best * (1.25f / 8388608.0f));
}

__global__ void k_zero(float* p) { *p = 0.0f; }

// ---------------------------------------------------------------------------
// Host-side orchestration
// ---------------------------------------------------------------------------
static inline unsigned cdiv(size_t a, size_t b) { return (unsigned)((a + b - 1) / b); }

extern "C" void kernel_launch(void* const* d_in, const int* in_sizes, int n_in,
                              void* d_out, int out_size, void* d_ws, size_t ws_size,
                              hipStream_t stream)
{
    const float* x       = (const float*)d_in[0];
    const float* enc_w1  = (const float*)d_in[1];
    const float* enc_b1  = (const float*)d_in[2];
    const float* enc_w2  = (const float*)d_in[3];
    const float* enc_b2  = (const float*)d_in[4];
    const float* er1_w3  = (const float*)d_in[5];
    const float* er1_w1  = (const float*)d_in[6];
    const float* er2_w3  = (const float*)d_in[7];
    const float* er2_w1  = (const float*)d_in[8];
    const float* codebook= (const float*)d_in[9];
    const float* dr1_w3  = (const float*)d_in[10];
    const float* dr1_w1  = (const float*)d_in[11];
    const float* dr2_w3  = (const float*)d_in[12];
    const float* dr2_w1  = (const float*)d_in[13];
    const float* dt1_w   = (const float*)d_in[14];
    const float* dt1_b   = (const float*)d_in[15];
    const float* dt2_w   = (const float*)d_in[16];
    const float* dt2_b   = (const float*)d_in[17];

    float* out  = (float*)d_out;
    float* loss = out + (out_size - 1);

    char* ws = (char*)d_ws;
    const size_t A_ELEMS = (size_t)32768 * 4096;  // == 131072*1024 bf16
    bf16_t* Abuf = (bf16_t*)(ws);
    bf16_t* Wbuf = (bf16_t*)(ws + A_ELEMS * 2);
    float*  T1   = (float*)(ws + A_ELEMS * 2 + (size_t)256 * 4096 * 2);
    float*  P0   = (float*)((char*)T1 + (size_t)131072 * 256 * 4);
    float*  P1   = (float*)((char*)P0 + (size_t)32768 * 256 * 4);
    (void)ws_size; (void)n_in; (void)in_sizes;

    const int TB = 256;
    auto gemm = [&](const bf16_t* A, const bf16_t* Bt, float* o,
                    const float* bias, const float* res,
                    int M, int N, int Kpad, int mode,
                    int Hq, int Wq, int Ho, int Wo, int ph, int pw, int CoutReal) {
        if ((M & 127) == 0 && (N & 127) == 0) {
            unsigned blocks = (unsigned)((M >> 7) * (N >> 7));
            k_gemm_tiled_wmma<<<blocks, TB, 0, stream>>>(A, Bt, o, bias, res,
                M, N, Kpad, mode, Hq, Wq, Ho, Wo, ph, pw, CoutReal);
        } else {
            unsigned blocks = cdiv((size_t)(M >> 4) * (N >> 4), 8);
            k_gemm_bf16_wmma<<<blocks, TB, 0, stream>>>(A, Bt, o, bias, res,
                M, N, Kpad, mode, Hq, Wq, Ho, Wo, ph, pw, CoutReal);
        }
    };

    k_zero<<<1, 1, 0, stream>>>(loss);

    // encoder conv1: 3->256, k4 s2 p1, 128->64
    {
        int M = 32 * 64 * 64, K = 3 * 16, Kp = 64;
        k_pack_w_conv<<<cdiv((size_t)256 * Kp, TB), TB, 0, stream>>>(enc_w1, Wbuf, 256, K, Kp, 256);
        k_im2col_conv<<<cdiv((size_t)M * Kp, TB), TB, 0, stream>>>(
            x, Abuf, 32, 3, 128, 128, 4, 4, 2, 1, 64, 64, K, Kp, 1, 0);
        gemm(Abuf, Wbuf, T1, enc_b1, nullptr, M, 256, Kp, 0, 0,0,0,0,0,0,0);
    }
    // encoder conv2: 256->256, k4 s2 p1, 64->32
    {
        int M = 32 * 32 * 32, K = 256 * 16, Kp = 4096;
        k_pack_w_conv<<<cdiv((size_t)256 * Kp, TB), TB, 0, stream>>>(enc_w2, Wbuf, 256, K, Kp, 256);
        k_im2col_conv<<<cdiv((size_t)M * Kp, TB), TB, 0, stream>>>(
            T1, Abuf, 32, 256, 64, 64, 4, 4, 2, 1, 32, 32, K, Kp, 0, 0);
        gemm(Abuf, Wbuf, P0, enc_b2, nullptr, M, 256, Kp, 0, 0,0,0,0,0,0,0);
    }
    // residual blocks
    auto res_block = [&](const float* w3, const float* w1) {
        int M = 32 * 32 * 32;
        int K3 = 256 * 9, Kp3 = 2304;
        k_pack_w_conv<<<cdiv((size_t)256 * Kp3, TB), TB, 0, stream>>>(w3, Wbuf, 256, K3, Kp3, 256);
        k_im2col_conv<<<cdiv((size_t)M * Kp3, TB), TB, 0, stream>>>(
            P0, Abuf, 32, 256, 32, 32, 3, 3, 1, 1, 32, 32, K3, Kp3, 0, 1);
        gemm(Abuf, Wbuf, P1, nullptr, nullptr, M, 256, Kp3, 0, 0,0,0,0,0,0,0);
        int K1 = 256, Kp1 = 256;
        k_pack_w_conv<<<cdiv((size_t)256 * Kp1, TB), TB, 0, stream>>>(w1, Wbuf, 256, K1, Kp1, 256);
        k_im2col_conv<<<cdiv((size_t)M * Kp1, TB), TB, 0, stream>>>(
            P1, Abuf, 32, 256, 32, 32, 1, 1, 1, 0, 32, 32, K1, Kp1, 0, 1);
        gemm(Abuf, Wbuf, P0, nullptr, P0, M, 256, Kp1, 0, 0,0,0,0,0,0,0);
    };
    res_block(er1_w3, er1_w1);
    res_block(er2_w3, er2_w1);

    // vector quantization (in-place on P0) + loss
    k_vq<<<512, 256, 0, stream>>>(P0, codebook, loss, 131072);

    res_block(dr1_w3, dr1_w1);
    res_block(dr2_w3, dr2_w1);

    // deconv1: 256->256, k4 s2 p1, 32->64, 4 parity classes -> T1
    for (int ph = 0; ph < 2; ++ph)
        for (int pw = 0; pw < 2; ++pw) {
            int M = 32 * 32 * 32, Kp = 256 * 4;
            k_pack_w_deconv<<<cdiv((size_t)256 * Kp, TB), TB, 0, stream>>>(
                dt1_w, Wbuf, 256, 256, 256, ph, pw);
            k_im2col_deconv<<<cdiv((size_t)M * Kp, TB), TB, 0, stream>>>(
                P0, Abuf, 32, 256, 32, 32, ph, pw);
            gemm(Abuf, Wbuf, T1, dt1_b, nullptr, M, 256, Kp,
                 1, 32, 32, 64, 64, ph, pw, 256);
        }

    // deconv2: 256->3, k4 s2 p1, 64->128, tanh, NCHW -> d_out
    for (int ph = 0; ph < 2; ++ph)
        for (int pw = 0; pw < 2; ++pw) {
            int M = 32 * 64 * 64, Kp = 256 * 4;
            k_pack_w_deconv<<<cdiv((size_t)16 * Kp, TB), TB, 0, stream>>>(
                dt2_w, Wbuf, 256, 3, 16, ph, pw);
            k_im2col_deconv<<<cdiv((size_t)M * Kp, TB), TB, 0, stream>>>(
                T1, Abuf, 32, 256, 64, 64, ph, pw);
            gemm(Abuf, Wbuf, out, dt2_b, nullptr, M, 16, Kp,
                 2, 64, 64, 128, 128, ph, pw, 3);
        }
}
